// SynthesisBlock_532575945157
// MI455X (gfx1250) — matrix operand order repaired and checked by reading, not compiled, not run on hardware
//
#include <hip/hip_runtime.h>
#include <hip/hip_bf16.h>

typedef __attribute__((ext_vector_type(16))) _Float16 v16h;
typedef __attribute__((ext_vector_type(8)))  float    v8f;

// ---------------------------------------------------------------------------
// Problem constants (from reference): N=8, Ci=256, Co=128, S=512, H=W=64,
// output spatial 128x128.  Activations stored f16 NHWC with 1-px zero halo:
// [N][130][130][C].
// ---------------------------------------------------------------------------
#define NB   8
#define CI1  256
#define CO   128
#define SDIM 512
#define HO   128
#define WO   128
#define HP   130   // halo pitch (128 + 2)

// ---------------------------------------------------------------------------
// m[n][c] = style[n] . mw[c] + mb[c] + 1
// ---------------------------------------------------------------------------
__global__ void style_mod_kernel(const float* __restrict__ style,
                                 const float* __restrict__ mw,
                                 const float* __restrict__ mb,
                                 float* __restrict__ m, int C) {
  int idx = blockIdx.x * 256 + threadIdx.x;
  if (idx >= NB * C) return;
  int n = idx / C, c = idx - n * C;
  const float* sp = style + (size_t)n * SDIM;
  const float* wp = mw + (size_t)c * SDIM;
  float acc = mb[c] + 1.0f;
  for (int s = 0; s < SDIM; ++s) acc = fmaf(sp[s], wp[s], acc);
  m[idx] = acc;
}

// wsq[co][ci] = sum over 3x3 of w^2
__global__ void wsq_kernel(const float* __restrict__ w, float* __restrict__ wsq,
                           int total) {
  int idx = blockIdx.x * 256 + threadIdx.x;
  if (idx >= total) return;
  const float* p = w + (size_t)idx * 9;
  float s = 0.f;
#pragma unroll
  for (int i = 0; i < 9; ++i) s = fmaf(p[i], p[i], s);
  wsq[idx] = s;
}

// demod[n][co] = rsqrt( sum_ci m[n][ci]^2 * wsq[co][ci] + eps )
__global__ void demod_kernel(const float* __restrict__ wsq,
                             const float* __restrict__ m,
                             float* __restrict__ dem, int Cin) {
  int idx = blockIdx.x * 256 + threadIdx.x;
  if (idx >= NB * CO) return;
  int n = idx >> 7, co = idx & (CO - 1);
  const float* wp = wsq + (size_t)co * Cin;
  const float* mp = m + (size_t)n * Cin;
  float s = 0.f;
  for (int c = 0; c < Cin; ++c) { float mv = mp[c]; s = fmaf(mv * mv, wp[c], s); }
  dem[idx] = rsqrtf(s + 1e-8f);
}

// ---------------------------------------------------------------------------
// Pack base weights into per-lane WMMA A-fragments (contiguous k = 16*(lane/16) + e, matching B loads).
// Layout: wA[ktile][mtile][lane][16 f16], ktile = tap*CIC + cichunk.
// ---------------------------------------------------------------------------
__global__ void pack_wA_kernel(const float* __restrict__ w, _Float16* __restrict__ wA,
                               int Cin, int CIC, int total) {
  int idx = blockIdx.x * 256 + threadIdx.x;
  if (idx >= total) return;
  int e    = idx & 15;
  int lane = (idx >> 4) & 31;
  int mt   = (idx >> 9) & 7;
  int t    = idx >> 12;
  int tap  = t / CIC, cc = t - tap * CIC;
  int co   = (mt << 4) + (lane & 15);
  int vg   = e >> 1;
  int k_in = ((lane >> 4) << 4) + e;
  int ci   = (cc << 5) + k_in;
  wA[idx] = (_Float16)w[((size_t)co * Cin + ci) * 9 + tap];
}

// ---------------------------------------------------------------------------
// Zero the 1-px halo ring of a [NB][130][130][Cin] f16 buffer.
// 516 ring pixels per image; block = Cin threads.
// ---------------------------------------------------------------------------
__global__ void zero_halo_kernel(_Float16* __restrict__ buf, int Cin) {
  int n = blockIdx.x / 516;
  int p = blockIdx.x - n * 516;
  int ci = threadIdx.x;
  int y, x;
  if (p < 130)      { y = 0;         x = p; }
  else if (p < 260) { y = 129;       x = p - 130; }
  else if (p < 388) { y = p - 259;   x = 0; }     // y in 1..128
  else              { y = p - 387;   x = 129; }   // y in 1..128
  buf[(((size_t)n * HP + y) * HP + x) * Cin + ci] = (_Float16)0.f;
}

// ---------------------------------------------------------------------------
// Bilinear 2x upsample (half-pixel centers, edge clamp), scale by m1[n][ci],
// store f16 NHWC-halo [n][oy+1][ox+1][ci].  One block per output pixel.
// ---------------------------------------------------------------------------
__global__ __launch_bounds__(256) void upsample_mod_kernel(
    const float* __restrict__ x, const float* __restrict__ m1,
    _Float16* __restrict__ xu) {
  int b = blockIdx.x;                       // n*16384 + oy*128 + ox
  int ci = threadIdx.x;                     // 0..255
  int n = b >> 14;
  int pix = b & 16383;
  int oy = pix >> 7, ox = pix & 127;
  float fy = oy * 0.5f - 0.25f, fx = ox * 0.5f - 0.25f;
  int y0 = (int)floorf(fy), x0 = (int)floorf(fx);
  float wy = fy - (float)y0, wx = fx - (float)x0;
  int ya = min(max(y0, 0), 63), yb = min(max(y0 + 1, 0), 63);
  int xa = min(max(x0, 0), 63), xb = min(max(x0 + 1, 0), 63);
  const float* xp = x + (((size_t)(n * CI1 + ci)) << 12);   // 64*64 plane
  float v00 = xp[(ya << 6) + xa], v01 = xp[(ya << 6) + xb];
  float v10 = xp[(yb << 6) + xa], v11 = xp[(yb << 6) + xb];
  float v = (1.f - wy) * ((1.f - wx) * v00 + wx * v01) +
            wy * ((1.f - wx) * v10 + wx * v11);
  v *= m1[(n << 8) + ci];
  xu[(((size_t)n * HP + (oy + 1)) * HP + (ox + 1)) * CI1 + ci] = (_Float16)v;
}

// ---------------------------------------------------------------------------
// Implicit-GEMM 3x3 conv via v_wmma_f32_16x16x32_f16.
//   xin : [N][130][130][Cin] f16, zero halo, pre-scaled by modulation
//   wA  : packed A fragments [9*CIC][8][32][16] f16
// Block = 256 threads = 8 waves; wave mt covers co rows [16mt,16mt+16);
// block covers 32 output pixels of one row.  K order: tap-major; with the
// halo every B load is an unconditional global_load_b128 at an immediate
// offset from a single per-lane base pointer.
// ---------------------------------------------------------------------------
template <int CIC, bool FINAL>
__global__ __launch_bounds__(256) void conv3x3_wmma_kernel(
    const _Float16* __restrict__ xin, const _Float16* __restrict__ wA,
    const float* __restrict__ demod, const float* __restrict__ nw,
    const float* __restrict__ noise, const float* __restrict__ postscale,
    _Float16* __restrict__ outh, float* __restrict__ outf) {
  constexpr int Cin = CIC * 32;
  constexpr int RP  = HP * Cin;           // row pitch (elements)
  const int lane = threadIdx.x & 31;
  const int mt   = threadIdx.x >> 5;
  const int n  = blockIdx.z;
  const int oy = blockIdx.y;
  const int x0 = blockIdx.x << 5;
  const int nlo = lane & 15;
  const int hi  = lane >> 4;

  v8f acc0 = {};
  v8f acc1 = {};

  // Per-lane base pointers at the (oy, x0+nlo[+16]) interior pixel.
  const _Float16* bp0 = xin + ((size_t)n * HP + (oy + 1)) * RP +
                        (size_t)(x0 + nlo + 1) * Cin + (hi << 4);
  const _Float16* bp1 = bp0 + (size_t)16 * Cin;
  const _Float16* ap  = wA + (((size_t)mt * 32 + lane) << 4);

#pragma unroll
  for (int tap = 0; tap < 9; ++tap) {
    const int ky = tap / 3 - 1;
    const int kx = tap - (tap / 3) * 3 - 1;
    const int tofs = (ky * HP + kx) * Cin;         // compile-time constant
#pragma unroll
    for (int cc = 0; cc < CIC; ++cc) {
      const int t = tap * CIC + cc;
      v16h a  = *(const v16h*)(ap + ((size_t)t << 12));
      v16h b0 = *(const v16h*)(bp0 + tofs + (cc << 5));
      v16h b1 = *(const v16h*)(bp1 + tofs + (cc << 5));
      acc0 = __builtin_amdgcn_wmma_f32_16x16x32_f16(false, a, false, b0,
                                                    (short)0, acc0, false, false);
      acc1 = __builtin_amdgcn_wmma_f32_16x16x32_f16(false, a, false, b1,
                                                    (short)0, acc1, false, false);
    }
  }

  // Epilogue.  Lane holds cols N = nlo (+16 for acc1), rows co0..co0+7.
  const int co0 = (mt << 4) + (hi << 3);
  float dm[8], nwv[8], ps[8];
#pragma unroll
  for (int r = 0; r < 8; ++r) {
    const int co = co0 + r;
    dm[r]  = demod[(n << 7) + co];
    nwv[r] = nw[co];
    if constexpr (!FINAL) ps[r] = postscale[(n << 7) + co];
  }
#pragma unroll
  for (int ct = 0; ct < 2; ++ct) {
    const int px = x0 + (ct << 4) + nlo;
    const float nz = noise[(((size_t)n) << 14) + (oy << 7) + px];
    v8f acc = ct ? acc1 : acc0;
    union { _Float16 h[8]; float4 f; } pk;
#pragma unroll
    for (int r = 0; r < 8; ++r) {
      const int co = co0 + r;
      float v = fmaf(nwv[r], nz, acc[r] * dm[r]);
      v = (v > 0.f) ? v : 0.2f * v;
      if constexpr (FINAL) {
        outf[((((size_t)n << 7) + co) << 14) + (oy << 7) + px] = v;
      } else {
        pk.h[r] = (_Float16)(v * ps[r]);
      }
    }
    if constexpr (!FINAL) {
      *(float4*)(outh + (((size_t)n * HP + (oy + 1)) * HP + (px + 1)) * CO + co0) = pk.f;
    }
  }
}

// ---------------------------------------------------------------------------
// toRGB: rgb[n][o][p] = sum_c out[n][c][p]*rgb_w[o][c] + rgb_b[o]
// ---------------------------------------------------------------------------
__global__ __launch_bounds__(256) void rgb_kernel(const float* __restrict__ outp,
                                                  const float* __restrict__ rw,
                                                  const float* __restrict__ rb,
                                                  float* __restrict__ rgb) {
  int idx = blockIdx.x * 256 + threadIdx.x;
  if (idx >= NB * HO * WO) return;
  int n = idx >> 14, pix = idx & 16383;
  float a0 = rb[0], a1 = rb[1], a2 = rb[2];
  const float* op = outp + (((size_t)n) << 21) + pix;
  for (int c = 0; c < CO; ++c) {
    float v = op[(size_t)c << 14];
    a0 = fmaf(v, rw[c], a0);
    a1 = fmaf(v, rw[128 + c], a1);
    a2 = fmaf(v, rw[256 + c], a2);
  }
  float* rp = rgb + (((size_t)n * 3) << 14) + pix;
  rp[0]           = a0;
  rp[1 << 14]     = a1;
  rp[2 << 14]     = a2;
}

// ---------------------------------------------------------------------------
extern "C" void kernel_launch(void* const* d_in, const int* in_sizes, int n_in,
                              void* d_out, int out_size, void* d_ws, size_t ws_size,
                              hipStream_t stream) {
  (void)in_sizes; (void)n_in; (void)out_size; (void)ws_size;
  const float* x      = (const float*)d_in[0];
  const float* style1 = (const float*)d_in[1];
  const float* style2 = (const float*)d_in[2];
  const float* noise  = (const float*)d_in[3];
  const float* w1     = (const float*)d_in[4];
  const float* mw1    = (const float*)d_in[5];
  const float* mb1    = (const float*)d_in[6];
  const float* nw1    = (const float*)d_in[7];
  const float* w2     = (const float*)d_in[8];
  const float* mw2    = (const float*)d_in[9];
  const float* mb2    = (const float*)d_in[10];
  const float* nw2    = (const float*)d_in[11];
  const float* rgbw   = (const float*)d_in[12];
  const float* rgbb   = (const float*)d_in[13];

  float* outp = (float*)d_out;                                   // [8,128,128,128]
  float* rgbp = outp + (size_t)NB * CO * HO * WO;                // [8,3,128,128]

  // Workspace carve-out (256B aligned slabs).
  char* wsp = (char*)d_ws;
  auto carve = [&](size_t bytes) -> void* {
    void* p = (void*)wsp;
    wsp += (bytes + 255) & ~(size_t)255;
    return p;
  };
  float*    m1     = (float*)   carve((size_t)NB * CI1 * 4);
  float*    m2     = (float*)   carve((size_t)NB * CO * 4);
  float*    dem1   = (float*)   carve((size_t)NB * CO * 4);
  float*    dem2   = (float*)   carve((size_t)NB * CO * 4);
  float*    wsq1   = (float*)   carve((size_t)CO * CI1 * 4);
  float*    wsq2   = (float*)   carve((size_t)CO * CO * 4);
  _Float16* whA1   = (_Float16*)carve((size_t)72 * 8 * 32 * 16 * 2);       // 576 KB
  _Float16* whA2   = (_Float16*)carve((size_t)36 * 8 * 32 * 16 * 2);       // 288 KB
  _Float16* xu     = (_Float16*)carve((size_t)NB * HP * HP * CI1 * 2);     // ~66 MB
  _Float16* out1h  = (_Float16*)carve((size_t)NB * HP * HP * CO * 2);      // ~33 MB

  // Modulation vectors (s+1) and demodulation factors.
  style_mod_kernel<<<(NB * CI1 + 255) / 256, 256, 0, stream>>>(style1, mw1, mb1, m1, CI1);
  style_mod_kernel<<<(NB * CO + 255) / 256, 256, 0, stream>>>(style2, mw2, mb2, m2, CO);
  wsq_kernel<<<(CO * CI1 + 255) / 256, 256, 0, stream>>>(w1, wsq1, CO * CI1);
  wsq_kernel<<<(CO * CO + 255) / 256, 256, 0, stream>>>(w2, wsq2, CO * CO);
  demod_kernel<<<(NB * CO + 255) / 256, 256, 0, stream>>>(wsq1, m1, dem1, CI1);
  demod_kernel<<<(NB * CO + 255) / 256, 256, 0, stream>>>(wsq2, m2, dem2, CO);

  // Pack weights into WMMA A-fragment layout (f16).
  pack_wA_kernel<<<(72 * 4096 + 255) / 256, 256, 0, stream>>>(w1, whA1, CI1, 8, 72 * 4096);
  pack_wA_kernel<<<(36 * 4096 + 255) / 256, 256, 0, stream>>>(w2, whA2, CO, 4, 36 * 4096);

  // Zero halos, then fill interiors.
  zero_halo_kernel<<<NB * 516, CI1, 0, stream>>>(xu, CI1);
  zero_halo_kernel<<<NB * 516, CO, 0, stream>>>(out1h, CO);

  // Upsample + channel modulation -> f16 NHWC (halo layout).
  upsample_mod_kernel<<<NB * HO * WO, 256, 0, stream>>>(x, m1, xu);

  // conv1 (256 -> 128), epilogue folds m2 for conv2's modulation.
  conv3x3_wmma_kernel<8, false><<<dim3(WO / 32, HO, NB), 256, 0, stream>>>(
      xu, whA1, dem1, nw1, noise, m2, out1h, nullptr);

  // conv2 (128 -> 128), final fp32 NCHW output.
  conv3x3_wmma_kernel<4, true><<<dim3(WO / 32, HO, NB), 256, 0, stream>>>(
      out1h, whA2, dem2, nw2, noise, nullptr, nullptr, outp);

  // toRGB 1x1.
  rgb_kernel<<<(NB * HO * WO + 255) / 256, 256, 0, stream>>>(outp, rgbw, rgbb, rgbp);
}
